// Model1_27625229648356
// MI455X (gfx1250) — matrix operand order repaired and checked by reading, not compiled, not run on hardware
//
#include <hip/hip_runtime.h>
#include <hip/hip_bf16.h>

typedef __attribute__((ext_vector_type(16))) _Float16 v16h;
typedef __attribute__((ext_vector_type(8)))  float    v8f;
typedef __attribute__((ext_vector_type(4)))  float    v4f;

// ---------------------------------------------------------------------------
// Conv4d (valid) + bias + ReLU as implicit GEMM on the WMMA pipe.
//   A (16xK): weights for one 16-row M tile, f16, pre-swizzled in LDS in exact
//             A-fragment lane order -> one 32B LDS read per fragment.
//   B (Kx16): im2col gather via a precomputed LDS offset table (affine im2col).
//   Per wave: TWO N tiles (32 output positions), K unrolled x2
//             => 4 v_wmma_f32_16x16x32_f16 per loop body, A shared across N.
//   CONTIG4 : K order is kw-fastest; when kW==4 (or K fully contiguous, fc1)
//             every 4 consecutive K are contiguous in memory and fragments are
//             16-aligned in K -> gather with global_load_b128 (4x fewer VMEM).
// ---------------------------------------------------------------------------

struct Pos { int inBase; int b, t0, d0, h0, w0; bool valid; };

__device__ __forceinline__ Pos decodePos(int n, int P, int Wo, int Ho, int Do, int To,
                                         int Cin, int T, int D, int H, int W)
{
    Pos p;
    p.valid = (n < P);
    int r = p.valid ? n : 0;
    p.w0 = r % Wo; r /= Wo;
    p.h0 = r % Ho; r /= Ho;
    p.d0 = r % Do; r /= Do;
    p.t0 = r % To; r /= To;
    p.b  = r;
    p.inBase = ((((p.b * Cin) * T + p.t0) * D + p.d0) * H + p.h0) * W + p.w0;
    return p;
}

union Frag { v16h v; _Float16 h[16]; };

// Build one B fragment (element e <-> K = kb + e, contiguous k = 16h + e order, same as A in LDS).
template <bool CONTIG4>
__device__ __forceinline__ void gatherB(const float* __restrict__ x, int inBase,
                                        const int* __restrict__ koff, int kb, Frag& f)
{
    if (CONTIG4) {
#pragma unroll
        for (int g = 0; g < 4; ++g) {
            int o = koff[kb + g * 4];
            v4f q = *(const v4f*)&x[inBase + o];      // dword-aligned b128 gather
#pragma unroll
            for (int j = 0; j < 4; ++j) f.h[g * 4 + j] = (_Float16)q[j];
        }
    } else {
#pragma unroll
        for (int e = 0; e < 16; ++e)
            f.h[e] = (_Float16)x[inBase + koff[kb + e]];
    }
}

template <bool CONTIG4>
__global__ void conv4d_wmma_f16(
    const float* __restrict__ x, const float* __restrict__ w,
    const float* __restrict__ bias, float* __restrict__ out,
    int Cin, int T, int D, int H, int W,
    int Cout, int kT, int kD, int kH, int kW,
    int To, int Do, int Ho, int Wo,
    int K, int Kpad, int P, int nTiles)
{
    extern __shared__ char smem[];
    _Float16* wFrag = (_Float16*)smem;                     // 16*Kpad f16 (A-fragment order)
    int*      koff  = (int*)(smem + 32 * (size_t)Kpad);    // Kpad im2col offsets
    float*    biasS = (float*)(smem + 36 * (size_t)Kpad);  // 16 floats

    const int tid   = threadIdx.x;
    const int mBase = blockIdx.y * 16;

    // --- im2col offset table (tail padded with 0 -> reads real finite data,
    //     the matching A rows are 0, so padded products are exactly 0)
    for (int k = tid; k < Kpad; k += blockDim.x) {
        int off = 0;
        if (k < K) {
            int r  = k;
            int kw = r % kW; r /= kW;
            int kh = r % kH; r /= kH;
            int kd = r % kD; r /= kD;
            int kt = r % kT; r /= kT;
            int ci = r;
            off = (((ci * T + kt) * D + kd) * H + kh) * W + kw;
        }
        koff[k] = off;
    }
    if (tid < 16) biasS[tid] = (mBase + tid < Cout) ? bias[mBase + tid] : 0.0f;

    // --- weights -> LDS in the 16x32 f16 A-fragment layout:
    // lane L (<16): M=L, K = {0..15}; lane L+16: M=L, K = {16..31}
    const int total = Kpad * 16;           // (Kpad/32) chunks * 32 lanes * 16 elems
    for (int idx = tid; idx < total; idx += blockDim.x) {
        int e     = idx & 15;
        int lane  = (idx >> 4) & 31;
        int chunk = idx >> 9;
        int m     = lane & 15;
        int kBase = (lane < 16) ? 0 : 16;
        int p     = e >> 1;
        int kk    = chunk * 32 + kBase + e + (p - p);
        float v = 0.0f;
        int mg = mBase + m;
        if (mg < Cout && kk < K) v = w[mg * K + kk];
        wFrag[idx] = (_Float16)v;
    }
    __syncthreads();

    const int wave = threadIdx.x >> 5;
    const int lane = threadIdx.x & 31;
    const int tile = blockIdx.x * (blockDim.x >> 5) + wave;
    if (tile >= nTiles) return;            // wave-uniform; no barriers after this

    // Two output positions per lane (two 16-wide N tiles per wave).
    const int n0 = tile * 32 + (lane & 15);
    const int n1 = n0 + 16;
    const Pos p0 = decodePos(n0, P, Wo, Ho, Do, To, Cin, T, D, H, W);
    const Pos p1 = decodePos(n1, P, Wo, Ho, Do, To, Cin, T, D, H, W);

    v8f acc0 = {}, acc1 = {};
    const int kBaseLane = (lane < 16) ? 0 : 16;
    const int nChunks   = Kpad >> 5;       // even (Kpad is a multiple of 64)

    for (int c = 0; c < nChunks; c += 2) {
        const int kb0 = (c + 0) * 32 + kBaseLane;
        const int kb1 = (c + 1) * 32 + kBaseLane;

        // Prefetch the next chunk pair's gather stream (global_prefetch_b8).
        if (c + 2 < nChunks)
            __builtin_prefetch(&x[p0.inBase + koff[kb0 + 64]], 0, 1);

        // A fragments: aligned 32-byte LDS reads (pre-swizzled).
        v16h a0 = *(const v16h*)&wFrag[((c + 0) * 32 + lane) * 16];
        v16h a1 = *(const v16h*)&wFrag[((c + 1) * 32 + lane) * 16];

        Frag b00, b01, b10, b11;
        gatherB<CONTIG4>(x, p0.inBase, koff, kb0, b00);
        gatherB<CONTIG4>(x, p1.inBase, koff, kb0, b01);
        gatherB<CONTIG4>(x, p0.inBase, koff, kb1, b10);
        gatherB<CONTIG4>(x, p1.inBase, koff, kb1, b11);

        acc0 = __builtin_amdgcn_wmma_f32_16x16x32_f16(false, a0, false, b00.v, (short)0, acc0, false, false);
        acc1 = __builtin_amdgcn_wmma_f32_16x16x32_f16(false, a0, false, b01.v, (short)0, acc1, false, false);
        acc0 = __builtin_amdgcn_wmma_f32_16x16x32_f16(false, a1, false, b10.v, (short)0, acc0, false, false);
        acc1 = __builtin_amdgcn_wmma_f32_16x16x32_f16(false, a1, false, b11.v, (short)0, acc1, false, false);
    }

    // Epilogue: bias + ReLU + scatter (C/D layout: lanes<16 -> M=r, else M=r+8).
    const int S     = To * Do * Ho * Wo;
    const int mHalf = (lane < 16) ? 0 : 8;
    if (p0.valid) {
        const int outBase = p0.b * Cout * S + (((p0.t0 * Do + p0.d0) * Ho + p0.h0) * Wo + p0.w0);
#pragma unroll
        for (int rr = 0; rr < 8; ++rr) {
            int m = mBase + mHalf + rr;
            if (m < Cout) {
                float v = acc0[rr] + biasS[mHalf + rr];
                out[outBase + m * S] = v > 0.0f ? v : 0.0f;
            }
        }
    }
    if (p1.valid) {
        const int outBase = p1.b * Cout * S + (((p1.t0 * Do + p1.d0) * Ho + p1.h0) * Wo + p1.w0);
#pragma unroll
        for (int rr = 0; rr < 8; ++rr) {
            int m = mBase + mHalf + rr;
            if (m < Cout) {
                float v = acc1[rr] + biasS[mHalf + rr];
                out[outBase + m * S] = v > 0.0f ? v : 0.0f;
            }
        }
    }
}

// fc2 (33 -> 1) + sigmoid: trivially latency bound, plain VALU.
__global__ void fc2_sigmoid(const float* __restrict__ h, const float* __restrict__ w,
                            const float* __restrict__ b, float* __restrict__ out,
                            int Bn, int Kn)
{
    int i = blockIdx.x * blockDim.x + threadIdx.x;
    if (i >= Bn) return;
    float s = b[0];
    for (int k = 0; k < Kn; ++k) s += h[i * Kn + k] * w[k];
    out[i] = 1.0f / (1.0f + expf(-s));
}

// ---------------------------------------------------------------------------
static void launch_conv(const float* x, const float* w, const float* bias, float* out,
                        int B, int Cin, int T, int D, int H, int W,
                        int Cout, int kT, int kD, int kH, int kW,
                        hipStream_t stream)
{
    const int To = T - kT + 1, Do = D - kD + 1, Ho = H - kH + 1, Wo = W - kW + 1;
    const int K    = Cin * kT * kD * kH * kW;
    const int Kpad = (K + 63) & ~63;                // multiple of 64 -> even #chunks
    const int P    = B * To * Do * Ho * Wo;
    const int nTiles = (P + 31) / 32;               // 32 output positions per wave
    const int wavesPerBlock = 8;                    // 256 threads, wave32
    const int blocks = (nTiles + wavesPerBlock - 1) / wavesPerBlock;
    const int mTiles = (Cout + 15) / 16;
    const size_t smem = (size_t)Kpad * 36 + 64;     // wFrag(32*Kpad) + koff(4*Kpad) + bias
    dim3 grid(blocks, mTiles);
    const bool contig4 = (kW == 4) || (kT * kD * kH * kW == 1);
    if (contig4)
        conv4d_wmma_f16<true><<<grid, 256, smem, stream>>>(
            x, w, bias, out, Cin, T, D, H, W, Cout, kT, kD, kH, kW,
            To, Do, Ho, Wo, K, Kpad, P, nTiles);
    else
        conv4d_wmma_f16<false><<<grid, 256, smem, stream>>>(
            x, w, bias, out, Cin, T, D, H, W, Cout, kT, kD, kH, kW,
            To, Do, Ho, Wo, K, Kpad, P, nTiles);
}

extern "C" void kernel_launch(void* const* d_in, const int* in_sizes, int n_in,
                              void* d_out, int out_size, void* d_ws, size_t ws_size,
                              hipStream_t stream)
{
    const float* x    = (const float*)d_in[0];
    const float* w1   = (const float*)d_in[1];  const float* b1 = (const float*)d_in[2];
    const float* w2   = (const float*)d_in[3];  const float* b2 = (const float*)d_in[4];
    const float* w3   = (const float*)d_in[5];  const float* b3 = (const float*)d_in[6];
    const float* w4   = (const float*)d_in[7];  const float* b4 = (const float*)d_in[8];
    const float* w5   = (const float*)d_in[9];  const float* b5 = (const float*)d_in[10];
    const float* fc1w = (const float*)d_in[11]; const float* fc1b = (const float*)d_in[12];
    const float* fc2w = (const float*)d_in[13]; const float* fc2b = (const float*)d_in[14];
    float* out = (float*)d_out;

    // Workspace ping-pong: slot A sized for h1 (largest), slot B for h2.
    const size_t slotA = (size_t)256 * 3 * 15 * 15 * 15 * 15;   // 38,880,000 floats
    float* bufA = (float*)d_ws;
    float* bufB = bufA + slotA;
    (void)ws_size; (void)in_sizes; (void)n_in; (void)out_size;

    // conv1: [256,1,18^4] -> [256,3,15^4]
    launch_conv(x,    w1, b1, bufA, 256, 1, 18, 18, 18, 18, 3, 4, 4, 4, 4, stream);
    // conv2: -> [256,3,12^4]
    launch_conv(bufA, w2, b2, bufB, 256, 3, 15, 15, 15, 15, 3, 4, 4, 4, 4, stream);
    // conv3: -> [256,4,9^4]
    launch_conv(bufB, w3, b3, bufA, 256, 3, 12, 12, 12, 12, 4, 4, 4, 4, 4, stream);
    // conv4: -> [256,5,6^4]
    launch_conv(bufA, w4, b4, bufB, 256, 4,  9,  9,  9,  9, 5, 4, 4, 4, 4, stream);
    // conv5: -> [256,5,4^4]  (flat = [256,1280]); kW=3 -> scalar gather path
    launch_conv(bufB, w5, b5, bufA, 256, 5,  6,  6,  6,  6, 5, 3, 3, 3, 3, stream);
    // fc1 as 1^4 conv: [256,1280] x [33,1280]^T -> [256,33], ReLU (contig K)
    launch_conv(bufA, fc1w, fc1b, bufB, 256, 1280, 1, 1, 1, 1, 33, 1, 1, 1, 1, stream);
    // fc2 + sigmoid -> [256,1]
    fc2_sigmoid<<<(256 + 255) / 256, 256, 0, stream>>>(bufB, fc2w, fc2b, out, 256, 33);
}
